// class_network_60120952209721
// MI455X (gfx1250) — matrix operand order repaired and checked by reading, not compiled, not run on hardware
//
#include <hip/hip_runtime.h>

// ---------------------------------------------------------------------------
// CDNA5 (gfx1250) feature detection
// ---------------------------------------------------------------------------
#if defined(__has_builtin)
#  if __has_builtin(__builtin_amdgcn_global_load_async_to_lds_b128)
#    define HAVE_ASYNC_LDS 1
#  endif
#endif
#ifndef HAVE_ASYNC_LDS
#  define HAVE_ASYNC_LDS 0
#endif

// Pointer types for the async-to-LDS builtin:
//   param0: global (AS1) int4*   param1: LDS (AS3) int4*
typedef int i32x4 __attribute__((ext_vector_type(4)));
typedef __attribute__((address_space(1))) i32x4 gi32x4;
typedef __attribute__((address_space(3))) i32x4 li32x4;

__device__ __forceinline__ void wait_async0() {
#if HAVE_ASYNC_LDS
#  if __has_builtin(__builtin_amdgcn_s_wait_asynccnt)
    __builtin_amdgcn_s_wait_asynccnt(0);
#  else
    asm volatile("s_wait_asynccnt 0x0" ::: "memory");
#  endif
#endif
}

// ---------------------------------------------------------------------------
// Types for CDNA5 WMMA (wave32): v_wmma_f32_16x16x32_bf16
// ---------------------------------------------------------------------------
typedef __attribute__((ext_vector_type(16))) __bf16 v16bf;
typedef __attribute__((ext_vector_type(8)))  float  v8f;

struct U8 { unsigned int u[8]; };

__device__ __forceinline__ unsigned short f2bf(float f) {
    unsigned int u = __builtin_bit_cast(unsigned int, f);
    unsigned int r = u + 0x7FFFu + ((u >> 16) & 1u);   // round-to-nearest-even
    return (unsigned short)(r >> 16);
}
__device__ __forceinline__ float bf2f(unsigned short b) {
    unsigned int u = ((unsigned int)b) << 16;
    return __builtin_bit_cast(float, u);
}

// ---------------------------------------------------------------------------
// Kernel 0a: f32 -> bf16 cast of network input (vectorized, 4 elems/thread)
// ---------------------------------------------------------------------------
__global__ __launch_bounds__(256) void cast_bf16(const float* __restrict__ x,
                                                 unsigned short* __restrict__ xb) {
    long i = ((long)blockIdx.x * 256 + threadIdx.x) * 4;
    float4 f = *(const float4*)&x[i];
    ushort4 o;
    o.x = f2bf(f.x); o.y = f2bf(f.y); o.z = f2bf(f.z); o.w = f2bf(f.w);
    *(ushort4*)&xb[i] = o;
}

// ---------------------------------------------------------------------------
// Kernel 0b: pre-pack conv weights f32 [Cout,Cin,K] -> bf16 [K][Cout][Cin].
// ---------------------------------------------------------------------------
__global__ __launch_bounds__(256) void pack_weights(const float* __restrict__ w,
                                                    unsigned short* __restrict__ wp,
                                                    int Cout, int Cin, int KW) {
    int i = blockIdx.x * 256 + threadIdx.x;
    if (i >= Cout * Cin * KW) return;
    int t  = i % KW;
    int r  = i / KW;
    int ci = r % Cin;
    int co = r / Cin;
    wp[((long)t * Cout + co) * Cin + ci] = f2bf(w[i]);
}

// ---------------------------------------------------------------------------
// Conv1d (same padding) as implicit GEMM with bf16 WMMA, f32 accumulate.
//   in : bf16 [B, CIN, L]     (raw bf16 bits)
//   wp : bf16 [KW][Cout][CIN] (pre-packed)
//   out: bf16 [B, Cout, L]
// Block = 256 threads (8 waves, 2(M) x 4(N)). Block tile 128(Cout) x 128(L).
// Wave tile 64 x 32 (4x2 WMMA 16x16x32 tiles). Double-buffered LDS staging;
// sA staged with GLOBAL_LOAD_ASYNC_TO_LDS_B128 when available.
// ci0/tap loops kept rolled (#pragma unroll 1) so the scheduler's lookahead
// stays inside the 256-VGPR direct window: no spills, no s_set_vgpr_msb,
// 4 waves/EU.
// ---------------------------------------------------------------------------
template <int CIN, int KW>
__global__ __launch_bounds__(256, 4) void conv_bn_wmma(const unsigned short* __restrict__ in,
                                                       const unsigned short* __restrict__ wp,
                                                       unsigned short* __restrict__ out,
                                                       int Cout) {
    constexpr int L     = 4096;
    constexpr int NT    = 128;
    constexpr int BCOLS = NT + KW - 1;
    constexpr int PAD   = KW / 2;
    constexpr int SAS   = 40;        // padded ci-stride of sA rows (80B, 16B-aligned chunks)
    constexpr int SBS   = 34;        // padded ci-stride of sB cols (68B -> stride-17 banks)
    constexpr int AROWS = KW * 128;
    constexpr int NS    = CIN / 32;

    __shared__ unsigned short sA[2][AROWS * SAS];
    __shared__ unsigned short sB[2][BCOLS * SBS];

    const int tid   = threadIdx.x;
    const int lane  = tid & 31;
    const int wv    = tid >> 5;
    const int waveM = wv >> 2;   // 0..1
    const int waveN = wv & 3;    // 0..3
    const int b     = blockIdx.z;
    const int n0    = blockIdx.x * NT;
    const int m0    = blockIdx.y * 128;
    const int grp   = lane >> 4;
    const int nn    = lane & 15;

    v8f acc[4][2] = {};

    // --- staging helpers -------------------------------------------------
    auto stageA = [&](int buf, int ci0) {
        // AROWS rows of 32 ci, 4 chunks of 8 ushorts (16B) each
        for (int i = tid; i < AROWS * 4; i += 256) {
            int  chunk = i & 3;
            int  r     = i >> 2;
            int  m     = r & 127;
            int  t     = r >> 7;
            long g     = ((long)(t * Cout + m0 + m)) * CIN + ci0 + chunk * 8;
            int  ld    = r * SAS + chunk * 8;
#if HAVE_ASYNC_LDS
            __builtin_amdgcn_global_load_async_to_lds_b128(
                (gi32x4*)(void*)(wp + g), (li32x4*)(void*)&sA[buf][ld], 0, 0);
#else
            *(uint4*)&sA[buf][ld] = *(const uint4*)&wp[g];
#endif
        }
    };
    auto stageB = [&](int buf, int ci0) {
        // transpose to [col][ci]; consecutive lanes read consecutive l (coalesced)
        for (int i = tid; i < 32 * BCOLS; i += 256) {
            int col = i % BCOLS;
            int ci  = i / BCOLS;
            int l   = n0 + col - PAD;
            unsigned short v = 0;
            if (l >= 0 && l < L) v = in[((long)(b * CIN + ci0 + ci)) * L + l];
            sB[buf][col * SBS + ci] = v;
        }
    };

    stageA(0, 0);
    stageB(0, 0);
    wait_async0();
    __syncthreads();

#pragma unroll 1
    for (int it = 0; it < NS; ++it) {
        const int cur = it & 1;
        if (it + 1 < NS) {               // prefetch next slab into other buffer
            stageA(cur ^ 1, (it + 1) * 32);
            stageB(cur ^ 1, (it + 1) * 32);
        }
        const unsigned short* A  = sA[cur];
        const unsigned short* Bm = sB[cur];

#pragma unroll 1
        for (int t = 0; t < KW; ++t) {
            // B fragments: lanes 0-15 hold K=0..7,16..23 (2/VGPR), lanes 16-31 K=8..15,24..31
            U8 bu0, bu1;
            const int c0 = waveN * 32 + nn + t;
#pragma unroll
            for (int v = 0; v < 8; ++v) {
                int kb   = ((v < 4) ? (2 * v) : (8 + 2 * v)) + 8 * grp;
                bu0.u[v] = *(const unsigned int*)&Bm[c0 * SBS + kb];
                bu1.u[v] = *(const unsigned int*)&Bm[(c0 + 16) * SBS + kb];
            }
            v16bf bf0 = __builtin_bit_cast(v16bf, bu0);
            v16bf bf1 = __builtin_bit_cast(v16bf, bu1);

#pragma unroll
            for (int ms = 0; ms < 4; ++ms) {
                U8 au;
                const int m = waveM * 64 + ms * 16 + nn;
#pragma unroll
                for (int v = 0; v < 8; ++v) {
                    int kb  = ((v < 4) ? (2 * v) : (8 + 2 * v)) + 8 * grp;
                    au.u[v] = *(const unsigned int*)&A[(t * 128 + m) * SAS + kb];
                }
                v16bf af = __builtin_bit_cast(v16bf, au);
                acc[ms][0] = __builtin_amdgcn_wmma_f32_16x16x32_bf16(
                    false, af, false, bf0, (short)0, acc[ms][0], false, false);
                acc[ms][1] = __builtin_amdgcn_wmma_f32_16x16x32_bf16(
                    false, af, false, bf1, (short)0, acc[ms][1], false, false);
            }
        }
        wait_async0();
        __syncthreads();
    }

    // C/D layout: VGPR r, lane -> N = lane&15, M = r + 8*(lane>>4)
#pragma unroll
    for (int ms = 0; ms < 4; ++ms)
#pragma unroll
        for (int ns = 0; ns < 2; ++ns)
#pragma unroll
            for (int r = 0; r < 8; ++r) {
                int co = m0 + waveM * 64 + ms * 16 + r + 8 * grp;
                int l  = n0 + waveN * 32 + ns * 16 + nn;
                out[((long)(b * Cout + co)) * L + l] = f2bf(acc[ms][ns][r]);
            }
}

// ---------------------------------------------------------------------------
// Per-channel BN statistics over (B, L): stats[c]=mean, stats[C+c]=rsqrt(var+eps)
// ---------------------------------------------------------------------------
__global__ __launch_bounds__(256) void bn_stats(const unsigned short* __restrict__ h,
                                                float* __restrict__ stats, int C) {
    const int c   = blockIdx.x;
    const int tid = threadIdx.x;
    float s = 0.f, q = 0.f;
    for (int b = 0; b < 32; ++b) {
        const unsigned short* p = h + ((long)(b * C + c)) * 4096;
        for (int i = tid * 8; i < 4096; i += 2048) {
            uint4 pk = *(const uint4*)&p[i];
            const unsigned short* e = (const unsigned short*)&pk;
#pragma unroll
            for (int j = 0; j < 8; ++j) {
                float v = bf2f(e[j]);
                s += v;
                q += v * v;
            }
        }
    }
    __shared__ float rs[256], rq[256];
    rs[tid] = s; rq[tid] = q;
    __syncthreads();
    for (int off = 128; off > 0; off >>= 1) {
        if (tid < off) { rs[tid] += rs[tid + off]; rq[tid] += rq[tid + off]; }
        __syncthreads();
    }
    if (tid == 0) {
        const float n    = 32.f * 4096.f;
        float       mean = rs[0] / n;
        float       var  = rq[0] / n - mean * mean;
        stats[c]     = mean;
        stats[C + c] = rsqrtf(fmaxf(var, 0.f) + 1e-5f);
    }
}

// ---------------------------------------------------------------------------
// In-place BN + ReLU (bf16 -> bf16), 8 elements/thread.
// ---------------------------------------------------------------------------
__global__ __launch_bounds__(256) void bn_relu(unsigned short* __restrict__ h,
                                               const float* __restrict__ stats,
                                               const float* __restrict__ gamma,
                                               const float* __restrict__ beta, int C) {
    long i = ((long)blockIdx.x * 256 + threadIdx.x) * 8;
    int  c = (int)((i >> 12) % C);    // 4096 elements per (b,c) row; 8 | 4096
    const float mean = stats[c], inv = stats[C + c], g = gamma[c], be = beta[c];
    uint4 pk = *(const uint4*)&h[i];
    unsigned short* e = (unsigned short*)&pk;
#pragma unroll
    for (int j = 0; j < 8; ++j) {
        float v = (bf2f(e[j]) - mean) * inv * g + be;
        e[j] = f2bf(fmaxf(v, 0.f));
    }
    *(uint4*)&h[i] = pk;
}

// ---------------------------------------------------------------------------
// Layer-3 fused BN + ReLU + mean-pool over L. Grid: (C=512, B=32).
// ---------------------------------------------------------------------------
__global__ __launch_bounds__(256) void bn_relu_pool(const unsigned short* __restrict__ h,
                                                    const float* __restrict__ stats,
                                                    const float* __restrict__ gamma,
                                                    const float* __restrict__ beta,
                                                    float* __restrict__ pooled) {
    const int C = 512;
    const int c = blockIdx.x, b = blockIdx.y, tid = threadIdx.x;
    const unsigned short* p = h + ((long)(b * C + c)) * 4096;
    const float mean = stats[c], inv = stats[C + c], g = gamma[c], be = beta[c];
    float s = 0.f;
    for (int i = tid * 8; i < 4096; i += 2048) {
        uint4 pk = *(const uint4*)&p[i];
        const unsigned short* e = (const unsigned short*)&pk;
#pragma unroll
        for (int j = 0; j < 8; ++j) {
            float v = (bf2f(e[j]) - mean) * inv * g + be;
            s += fmaxf(v, 0.f);
        }
    }
    __shared__ float rs[256];
    rs[tid] = s;
    __syncthreads();
    for (int off = 128; off > 0; off >>= 1) {
        if (tid < off) rs[tid] += rs[tid + off];
        __syncthreads();
    }
    if (tid == 0) pooled[b * C + c] = rs[0] * (1.f / 4096.f);
}

// ---------------------------------------------------------------------------
// Head: FC + gate softmax + top-3 + expert mix + balance loss. One block.
// out layout: [0 .. 4095] = final[32][128], [4096] = balance_loss
// ---------------------------------------------------------------------------
__global__ __launch_bounds__(256) void head(const float* __restrict__ pooled,
                                            const float* __restrict__ w_fc,
                                            const float* __restrict__ b_fc,
                                            const float* __restrict__ w_gate,
                                            const float* __restrict__ b_gate,
                                            const float* __restrict__ w_exp,
                                            const float* __restrict__ b_exp,
                                            float* __restrict__ out) {
    const int tid = threadIdx.x;
    __shared__ float sOut[32 * 192];
    __shared__ float sProb[32 * 5];
    __shared__ float sTopV[32 * 3];
    __shared__ int   sTopI[32 * 3];
    __shared__ float sMu[5];

    // FC: relu(pooled @ w_fc^T + b_fc)   [32 x 192], reduce over 512
    for (int i = tid; i < 32 * 192; i += 256) {
        int   bi = i / 192, d = i % 192;
        float a  = b_fc[d];
        const float* pr = pooled + bi * 512;
        const float* wr = w_fc + d * 512;
        for (int k = 0; k < 512; ++k) a += pr[k] * wr[k];
        sOut[i] = fmaxf(a, 0.f);
    }
    __syncthreads();

    // gate logits [32 x 5]
    for (int i = tid; i < 32 * 5; i += 256) {
        int   bi = i / 5, e = i % 5;
        float a  = b_gate[e];
        const float* orow = sOut + bi * 192;
        const float* wr   = w_gate + e * 192;
        for (int d = 0; d < 192; ++d) a += orow[d] * wr[d];
        sProb[i] = a;
    }
    __syncthreads();

    // softmax + top-3 per row (strict > picks lowest index on ties, like top_k)
    if (tid < 32) {
        int   bi = tid;
        float mx = sProb[bi * 5];
        for (int e = 1; e < 5; ++e) mx = fmaxf(mx, sProb[bi * 5 + e]);
        float p[5], sum = 0.f;
        for (int e = 0; e < 5; ++e) { p[e] = expf(sProb[bi * 5 + e] - mx); sum += p[e]; }
        float rinv = 1.f / sum;
        for (int e = 0; e < 5; ++e) { p[e] *= rinv; sProb[bi * 5 + e] = p[e]; }
        bool used[5] = {false, false, false, false, false};
        for (int k = 0; k < 3; ++k) {
            int   best = 0;
            float bv   = -1.f;
            for (int e = 0; e < 5; ++e)
                if (!used[e] && p[e] > bv) { bv = p[e]; best = e; }
            used[best]        = true;
            sTopV[bi * 3 + k] = bv;
            sTopI[bi * 3 + k] = best;
        }
    }
    __syncthreads();

    // mu + balance loss
    if (tid < 5) {
        float m = 0.f;
        for (int bi = 0; bi < 32; ++bi) m += sProb[bi * 5 + tid];
        sMu[tid] = m * (1.f / 32.f);
    }
    __syncthreads();
    if (tid == 0) {
        float ent = 0.f, var = 0.f;
        for (int e = 0; e < 5; ++e) {
            float mu = sMu[e];
            ent -= mu * logf(mu + 1e-10f);
            float d = mu - 0.2f;
            var += d * d;
        }
        out[32 * 128] = var + 0.1f * ent;
    }

    // final[b][nc] = sum_k topv * (sOut[b] . w_exp[e][nc] + b_exp[e][nc])
    for (int i = tid; i < 32 * 128; i += 256) {
        int   bi = i / 128, nc = i % 128;
        float accv = 0.f;
        const float* orow = sOut + bi * 192;
        for (int k = 0; k < 3; ++k) {
            int   e  = sTopI[bi * 3 + k];
            float tv = sTopV[bi * 3 + k];
            float a  = b_exp[e * 128 + nc];
            const float* we = w_exp + ((long)(e * 128 + nc)) * 192;
            for (int d = 0; d < 192; ++d) a += orow[d] * we[d];
            accv += tv * a;
        }
        out[i] = accv;
    }
}

// ---------------------------------------------------------------------------
// Launcher
// ---------------------------------------------------------------------------
extern "C" void kernel_launch(void* const* d_in, const int* in_sizes, int n_in,
                              void* d_out, int out_size, void* d_ws, size_t ws_size,
                              hipStream_t stream) {
    (void)in_sizes; (void)n_in; (void)out_size; (void)ws_size;

    const float* x      = (const float*)d_in[0];
    const float* w_bb   = (const float*)d_in[1];
    const float* g_bb   = (const float*)d_in[2];
    const float* b_bb   = (const float*)d_in[3];
    const float* w1     = (const float*)d_in[4];
    const float* g1     = (const float*)d_in[5];
    const float* b1     = (const float*)d_in[6];
    const float* w2     = (const float*)d_in[7];
    const float* g2     = (const float*)d_in[8];
    const float* b2     = (const float*)d_in[9];
    const float* w_fc   = (const float*)d_in[10];
    const float* b_fc   = (const float*)d_in[11];
    const float* w_exp  = (const float*)d_in[12];
    const float* b_exp  = (const float*)d_in[13];
    const float* w_gate = (const float*)d_in[14];
    const float* b_gate = (const float*)d_in[15];
    float*       out    = (float*)d_out;

    char* ws = (char*)d_ws;
    unsigned short* xb     = (unsigned short*)(ws);                 // 16 MB
    unsigned short* h1     = (unsigned short*)(ws + 16777216L);     // 32 MB
    unsigned short* h2     = (unsigned short*)(ws + 50331648L);     // 64 MB
    unsigned short* h3     = (unsigned short*)(ws + 117440512L);    // 128 MB
    float*          stats1 = (float*)(ws + 251658240L);
    float*          stats2 = (float*)(ws + 251659264L);
    float*          stats3 = (float*)(ws + 251661312L);
    float*          pooled = (float*)(ws + 251665408L);             // 32*512 f32
    unsigned short* wp1    = (unsigned short*)(ws + 251730944L);    // 5*128*64  bf16
    unsigned short* wp2    = (unsigned short*)(ws + 251812864L);    // 3*256*128 bf16
    unsigned short* wp3    = (unsigned short*)(ws + 252009472L);    // 3*512*256 bf16

    // 0) pre-pack weights + cast input to bf16
    pack_weights<<<160, 256, 0, stream>>>(w_bb, wp1, 128, 64, 5);
    pack_weights<<<384, 256, 0, stream>>>(w1, wp2, 256, 128, 3);
    pack_weights<<<1536, 256, 0, stream>>>(w2, wp3, 512, 256, 3);
    cast_bf16<<<8192, 256, 0, stream>>>(x, xb);      // 32*64*4096 / 1024

    // 1) conv1 64->128 K=5 + BN + ReLU
    conv_bn_wmma<64, 5><<<dim3(32, 1, 32), 256, 0, stream>>>(xb, wp1, h1, 128);
    bn_stats<<<128, 256, 0, stream>>>(h1, stats1, 128);
    bn_relu<<<8192, 256, 0, stream>>>(h1, stats1, g_bb, b_bb, 128);   // 16.7M/2048

    // 2) conv2 128->256 K=3 + BN + ReLU
    conv_bn_wmma<128, 3><<<dim3(32, 2, 32), 256, 0, stream>>>(h1, wp2, h2, 256);
    bn_stats<<<256, 256, 0, stream>>>(h2, stats2, 256);
    bn_relu<<<16384, 256, 0, stream>>>(h2, stats2, g1, b1, 256);      // 33.5M/2048

    // 3) conv3 256->512 K=3 + BN + ReLU + pool (pool fused)
    conv_bn_wmma<256, 3><<<dim3(32, 4, 32), 256, 0, stream>>>(h2, wp3, h3, 512);
    bn_stats<<<512, 256, 0, stream>>>(h3, stats3, 512);
    bn_relu_pool<<<dim3(512, 32), 256, 0, stream>>>(h3, stats3, g2, b2, pooled);

    // 4) head
    head<<<1, 256, 0, stream>>>(pooled, w_fc, b_fc, w_gate, b_gate, w_exp, b_exp, out);
}
